// FeatureAggregation_47373489275423
// MI455X (gfx1250) — matrix operand order repaired and checked by reading, not compiled, not run on hardware
//
#include <hip/hip_runtime.h>
#include <hip/hip_bf16.h>

typedef __attribute__((ext_vector_type(16))) _Float16 v16h;
typedef __attribute__((ext_vector_type(8)))  _Float16 v8h;
typedef __attribute__((ext_vector_type(8)))  float    v8f;

namespace {
constexpr int kC  = 256;   // channels of Z
constexpr int kHW = 256;   // Z spatial
constexpr int kSQ = 2048;  // queries
constexpr int kSK = 64;    // keys per query
constexpr int kE  = 512;   // embed dim
constexpr int kNH = 8;     // heads (num_heads input is fixed at 8)

// workspace layout, in _Float16 elements (~42.5 MB total)
constexpr size_t OFF_W    = 0;                          // in_proj_w f16   (1536x512)
constexpr size_t OFF_WKT  = OFF_W    + 1536ull*512;     // Wk^T f16 [e][d] (512x512)
constexpr size_t OFF_OUTW = OFF_WKT  + 512ull*512;      // out_proj_w f16  (512x512)
constexpr size_t OFF_QRY  = OFF_OUTW + 512ull*512;      // query f16       (2048x512)
constexpr size_t OFF_QL   = OFF_QRY  + 2048ull*512;     // ql f16          (2048x512)
constexpr size_t OFF_U    = OFF_QL   + 2048ull*512;     // u f16           (2048x8x512)
constexpr size_t OFF_CTX  = OFF_U    + 2048ull*8*512;   // ctx f16         (2048x8x512)
constexpr size_t OFF_AO   = OFF_CTX  + 2048ull*8*512;   // attn out f16    (2048x512)
} // namespace

// ---------------- CDNA5 async Global->LDS copy (ASYNCcnt path) ------------

__device__ inline void async_copy_b128(void* lds_dst, const void* gsrc) {
  // GLOBAL_LOAD_ASYNC_TO_LDS_B128: VDST = per-lane LDS byte address,
  // VADDR = per-lane 64-bit global address (GV mode, SADDR=off).
  uint32_t lds_addr = (uint32_t)(uintptr_t)lds_dst;
  asm volatile("global_load_async_to_lds_b128 %0, %1, off"
               :: "v"(lds_addr), "v"(gsrc) : "memory");
}

__device__ inline void wait_asynccnt0() {
  asm volatile("s_wait_asynccnt 0x0" ::: "memory");
}

// ---------------- WMMA operand loaders (CDNA5 16x16x32 f16 layouts) -------

__device__ inline v16h cat8(v8h lo, v8h hi) {
  v16h r;
#pragma unroll
  for (int j = 0; j < 8; ++j) { r[j] = lo[j]; r[8 + j] = hi[j]; }
  return r;
}

__device__ inline v16h zero16h() {
  v16h r;
#pragma unroll
  for (int j = 0; j < 16; ++j) r[j] = (_Float16)0.f;
  return r;
}

// A matrix 16x32: lane<16 -> row m=lane, K {k0..k0+7, k0+16..k0+23};
//                 lane>=16 -> row m=lane-16, K {k0+8..k0+15, k0+24..k0+31}
__device__ inline v16h load_A(const _Float16* Abase, int lda, int k0, int lane) {
  const int m  = lane & 15;
  const int kb = (lane < 16) ? 0 : 8;
  const _Float16* p = Abase + (size_t)m * lda + (k0 + kb);
  return cat8(*(const v8h*)p, *(const v8h*)(p + 16));
}

// A with only the first 8 rows defined (rows 8..15 zero)
__device__ inline v16h load_A_rows8(const _Float16* Abase, int lda, int k0, int lane) {
  const int m = lane & 15;
  if (m >= 8) return zero16h();
  const int kb = (lane < 16) ? 0 : 8;
  const _Float16* p = Abase + (size_t)m * lda + (k0 + kb);
  return cat8(*(const v8h*)p, *(const v8h*)(p + 16));
}

// B matrix 32x16 from Bt stored row-major [N][K]:
// lane<16 -> col n=lane, K {k0..k0+7, k0+16..k0+23}; lane>=16 -> col n=lane-16, K {k0+8..k0+15, k0+24..k0+31}
__device__ inline v16h load_Bt(const _Float16* Btbase, int ldb, int k0, int lane) {
  const int n  = lane & 15;
  const int kb = (lane < 16) ? 0 : 8;
  const _Float16* p = Btbase + (size_t)n * ldb + (k0 + kb);
  return cat8(*(const v8h*)p, *(const v8h*)(p + 16));
}

// B matrix 32x16 from B stored row-major [K][N'] (strided per-lane gather; LDS use)
__device__ inline v16h load_B_cols(const _Float16* B, int ldb, int n0, int k0, int lane) {
  const int n  = lane & 15;
  const int kb = (lane < 16) ? 0 : 8;
  v16h r;
#pragma unroll
  for (int j = 0; j < 16; ++j) r[j] = B[(size_t)(k0 + kb + (j < 8 ? j : j + 8)) * ldb + n0 + n];
  return r;
}

// ---------------- prep: f16 weight conversion, query build, q_z copy ------

__global__ __launch_bounds__(256) void prep_kernel(
    const float* __restrict__ ipw, const float* __restrict__ opw,
    const float* __restrict__ q_z, const float* __restrict__ Qm,
    const float* __restrict__ w2, const float* __restrict__ b2,
    _Float16* __restrict__ W16, _Float16* __restrict__ WkT,
    _Float16* __restrict__ OutW, _Float16* __restrict__ qry,
    float* __restrict__ out) {
  int idx = blockIdx.x * 256 + threadIdx.x;
  if (idx < 786432) {                       // in_proj_w -> f16
    W16[idx] = (_Float16)ipw[idx];
    return;
  }
  idx -= 786432;
  if (idx < 262144) {                       // WkT[e][d] = in_proj_w[512+d][e]
    int d = idx & 511, e = idx >> 9;
    WkT[idx] = (_Float16)ipw[(size_t)(512 + d) * 512 + e];
    return;
  }
  idx -= 262144;
  if (idx < 262144) {                       // out_proj_w -> f16
    OutW[idx] = (_Float16)opw[idx];
    return;
  }
  idx -= 262144;
  if (idx < 1048576) {                      // query = [q_z, Q@w2^T + b2]
    int q = idx >> 9, c = idx & 511;
    float v;
    if (c < 256) {
      v = q_z[(size_t)q * 256 + c];
    } else {
      int j = c - 256;
      v = Qm[q * 3 + 0] * w2[j * 3 + 0] + Qm[q * 3 + 1] * w2[j * 3 + 1] +
          Qm[q * 3 + 2] * w2[j * 3 + 2] + b2[j];
    }
    qry[idx] = (_Float16)v;
    return;
  }
  idx -= 1048576;
  {                                         // d_out[:, 0:256] = q_z (f32)
    int q = idx >> 8, c = idx & 255;
    out[(size_t)q * 768 + c] = q_z[(size_t)q * 256 + c];
  }
}

// ---------------- generic batched WMMA GEMM: C = A @ Bt^T (+bias) ---------
// A row-major [M][K] f16 (lda), Bt row-major [N][K] f16 (ldb).
// All 8 waves of a block share one 64-col B chunk, staged ONCE into LDS via
// async global->LDS b128 copies (ASYNCcnt); each wave owns one 16-row M tile.
// Grid: x = (M/128) * (N/64), y = batch. M % 128 == 0, N % 64 == 0, K % 32 == 0.

__global__ __launch_bounds__(256) void wmma_gemm(
    const _Float16* __restrict__ A, int lda, long sA,
    const _Float16* __restrict__ Bt, int ldb, long sB,
    const float* __restrict__ bias, int sBias, int hasBias,
    _Float16* __restrict__ C16, float* __restrict__ C32, int ldc, long sC,
    int M, int N, int K) {
  __shared__ __align__(16) _Float16 Bs[64 * 512];   // 64 KB, rows packed with ld = K

  const int b = blockIdx.y;
  A  += (size_t)b * sA;
  Bt += (size_t)b * sB;
  if (hasBias) bias += (size_t)b * sBias;
  if (C16) C16 += (size_t)b * sC;
  if (C32) C32 += (size_t)b * sC;

  const int lane = threadIdx.x & 31;
  const int wave = threadIdx.x >> 5;
  const int mtiles = M >> 4;
  const int blocksPerNc = mtiles >> 3;            // 8 M-tiles per block
  const int nc = blockIdx.x / blocksPerNc;        // 64-col chunk (shared by block)
  const int mt = (blockIdx.x % blocksPerNc) * 8 + wave;

  // ---- stage Bt[nc*64 .. nc*64+63][0..K) into LDS (async, 16B chunks) ----
  {
    const _Float16* Bchunk = Bt + (size_t)(nc * 64) * ldb;
    const int chunksPerRow = K >> 3;              // 8 halfs = 16B per chunk
    const int total = 64 * chunksPerRow;
    for (int t = threadIdx.x; t < total; t += 256) {
      const int row = t / chunksPerRow;
      const int off = (t - row * chunksPerRow) * 8;
      async_copy_b128(&Bs[row * K + off], Bchunk + (size_t)row * ldb + off);
    }
    wait_asynccnt0();
    __syncthreads();
  }

  const _Float16* Arow = A + (size_t)(mt * 16) * lda;
  v8f acc[4] = {};
  for (int k0 = 0; k0 < K; k0 += 32) {
    v16h a = load_A(Arow, lda, k0, lane);
#pragma unroll
    for (int t = 0; t < 4; ++t) {
      v16h bm = load_Bt(&Bs[(t * 16) * K], K, k0, lane);
      acc[t] = __builtin_amdgcn_wmma_f32_16x16x32_f16(
          false, a, false, bm, (short)0, acc[t], false, false);
    }
  }
  const int ncol  = lane & 15;
  const int mbase = (lane < 16) ? 0 : 8;
#pragma unroll
  for (int t = 0; t < 4; ++t) {
#pragma unroll
    for (int v = 0; v < 8; ++v) {
      const int row = mt * 16 + mbase + v;
      const int col = nc * 64 + t * 16 + ncol;
      float val = acc[t][v];
      if (hasBias) val += bias[col];
      if (C32) C32[(size_t)row * ldc + col] = val;
      else     C16[(size_t)row * ldc + col] = (_Float16)val;
    }
  }
}

// ---------------- fused per-query kernel ----------------------------------
// Builds key_q (64x512 f16) in LDS from bilinear gathers of Z (L2-resident)
// plus the tiny r-projection, then:
//   scores = u_q (8x512, padded to 16) @ key_q^T   (WMMA, waves 0..3)
//   softmax over the 64 keys (scale 1/sqrt(64))
//   ctx    = attn (16x64) @ key_q (64x512)         (WMMA, 8 waves x 4 tiles)

__global__ __launch_bounds__(256) void attn_kernel(
    const float* __restrict__ Z, const float* __restrict__ r,
    const float* __restrict__ w1, const float* __restrict__ b1,
    const _Float16* __restrict__ u, _Float16* __restrict__ ctx) {
  __shared__ __align__(16) _Float16 key[kSK][kE];     // 64 KB
  __shared__ int   cx[kSK], cy[kSK];
  __shared__ float crx[kSK], cry[kSK], cdx[kSK], cdy[kSK];
  __shared__ float s[16][64];
  __shared__ __align__(16) _Float16 a16[16][64];

  const int q    = blockIdx.x;
  const int tid  = threadIdx.x;
  const int lane = tid & 31;
  const int wave = tid >> 5;

  if (tid < kSK) {
    float x = r[((size_t)q * kSK + tid) * 2 + 0];
    float y = r[((size_t)q * kSK + tid) * 2 + 1];
    int x1 = (int)x, y1 = (int)y;
    cx[tid] = x1; cy[tid] = y1;
    crx[tid] = x; cry[tid] = y;
    cdx[tid] = x - (float)x1; cdy[tid] = y - (float)y1;
  }
  __syncthreads();

  // key[:, 0:256] = bilinear interp of Z (4 corner loads, hot in L2)
  for (int idx = tid; idx < kSK * kC; idx += 256) {
    int k = idx >> 8, c = idx & (kC - 1);
    const float* zp = Z + ((size_t)c * kHW + cx[k]) * kHW + cy[k];
    float dx = cdx[k], dy = cdy[k];
    float q11 = zp[0], q12 = zp[1], q21 = zp[kHW], q22 = zp[kHW + 1];
    float r1 = q11 * (1.f - dx) + q21 * dx;
    float r2 = q12 * (1.f - dx) + q22 * dx;
    key[k][c] = (_Float16)(r1 * (1.f - dy) + r2 * dy);
  }
  // key[:, 256:512] = r @ w1^T + b1
  for (int idx = tid; idx < kSK * 256; idx += 256) {
    int k = idx >> 8, h = idx & 255;
    key[k][256 + h] =
        (_Float16)(crx[k] * w1[h * 2 + 0] + cry[k] * w1[h * 2 + 1] + b1[h]);
  }
  __syncthreads();

  // scores: waves 0..3 each handle 16 keys; A rows = heads (rows 8..15 zero)
  if (wave < 4) {
    v8f acc = {};
    const _Float16* Aq = u + (size_t)q * (kNH * kE);
    for (int k0 = 0; k0 < kE; k0 += 32) {
      v16h a  = load_A_rows8(Aq, kE, k0, lane);
      v16h bm = load_Bt(&key[wave * 16][0], kE, k0, lane);
      acc = __builtin_amdgcn_wmma_f32_16x16x32_f16(
          false, a, false, bm, (short)0, acc, false, false);
    }
    const int mbase = (lane < 16) ? 0 : 8;
#pragma unroll
    for (int v = 0; v < 8; ++v)
      s[mbase + v][wave * 16 + (lane & 15)] = acc[v];
  }
  __syncthreads();

  // softmax per head over 64 keys (scale 1/sqrt(hd)=0.125)
  if (tid < kNH) {
    float mx = -1e30f;
    for (int k = 0; k < kSK; ++k) mx = fmaxf(mx, s[tid][k] * 0.125f);
    float sum = 0.f;
    for (int k = 0; k < kSK; ++k) sum += __expf(s[tid][k] * 0.125f - mx);
    float inv = 1.f / sum;
    for (int k = 0; k < kSK; ++k)
      a16[tid][k] = (_Float16)(__expf(s[tid][k] * 0.125f - mx) * inv);
  } else if (tid < 16) {
    for (int k = 0; k < kSK; ++k) a16[tid][k] = (_Float16)0.f;
  }
  __syncthreads();

  // ctx[h][e] = sum_k attn[h][k] * key[k][e]; wave w -> e-tiles 4w..4w+3
  {
    v8f acc[4] = {};
    for (int k0 = 0; k0 < kSK; k0 += 32) {
      v16h a = load_A(&a16[0][0], 64, k0, lane);
#pragma unroll
      for (int i = 0; i < 4; ++i) {
        v16h bm = load_B_cols(&key[0][0], kE, (wave * 4 + i) * 16, k0, lane);
        acc[i] = __builtin_amdgcn_wmma_f32_16x16x32_f16(
            false, a, false, bm, (short)0, acc[i], false, false);
      }
    }
    if (lane < 16) {  // rows 0..7 of the C tile = the 8 heads
      const int n = lane;
#pragma unroll
      for (int i = 0; i < 4; ++i)
#pragma unroll
        for (int v = 0; v < 8; ++v)
          ctx[(size_t)q * (kNH * kE) + (size_t)v * kE + (wave * 4 + i) * 16 + n] =
              (_Float16)acc[i][v];
    }
  }
}

// ---------------- host-side launcher --------------------------------------

extern "C" void kernel_launch(void* const* d_in, const int* in_sizes, int n_in,
                              void* d_out, int out_size, void* d_ws, size_t ws_size,
                              hipStream_t stream) {
  (void)in_sizes; (void)n_in; (void)out_size; (void)ws_size;
  const float* Z   = (const float*)d_in[0];
  const float* Qm  = (const float*)d_in[1];
  const float* q_z = (const float*)d_in[2];
  const float* r   = (const float*)d_in[3];
  const float* w1  = (const float*)d_in[4];
  const float* b1  = (const float*)d_in[5];
  const float* w2  = (const float*)d_in[6];
  const float* b2  = (const float*)d_in[7];
  const float* ipw = (const float*)d_in[8];
  const float* ipb = (const float*)d_in[9];
  const float* opw = (const float*)d_in[10];
  const float* opb = (const float*)d_in[11];
  // d_in[12] = num_heads (fixed 8)
  float* out   = (float*)d_out;
  _Float16* ws = (_Float16*)d_ws;

  _Float16* W16  = ws + OFF_W;
  _Float16* WkT  = ws + OFF_WKT;
  _Float16* OutW = ws + OFF_OUTW;
  _Float16* qry  = ws + OFF_QRY;
  _Float16* ql   = ws + OFF_QL;
  _Float16* u    = ws + OFF_U;
  _Float16* ctx  = ws + OFF_CTX;
  _Float16* ao   = ws + OFF_AO;

  // total prep elements = 786432+262144+262144+1048576+524288 = 2883584 = 11264*256
  prep_kernel<<<11264, 256, 0, stream>>>(ipw, opw, q_z, Qm, w2, b2,
                                         W16, WkT, OutW, qry, out);

  // ql = query @ Wq^T + bq          : (2048x512)@(512x512)
  wmma_gemm<<<dim3(128, 1), 256, 0, stream>>>(
      qry, 512, 0, W16, 512, 0, ipb, 0, 1,
      ql, nullptr, 512, 0, kSQ, 512, 512);

  // u_h = ql_h @ Wk_h               : batch 8, (2048x64)@(64x512)
  wmma_gemm<<<dim3(128, 8), 256, 0, stream>>>(
      ql, 512, 64, WkT, 512, 64, nullptr, 0, 0,
      u, nullptr, 4096, 512, kSQ, 512, 64);

  // fused key-build + scores + softmax + ctx, one WG per query
  attn_kernel<<<kSQ, 256, 0, stream>>>(Z, r, w1, b1, u, ctx);

  // out_h = ctx_h @ Wv_h^T + bv_h   : batch 8, (2048x512)@(512x64)
  wmma_gemm<<<dim3(16, 8), 256, 0, stream>>>(
      ctx, 4096, 512, W16 + (size_t)1024 * 512, 512, (long)64 * 512,
      ipb + 1024, 64, 1,
      ao, nullptr, 512, 64, kSQ, 64, 512);

  // final = ao @ out_proj_w^T + ob  -> d_out cols 256..767 (f32)
  wmma_gemm<<<dim3(128, 1), 256, 0, stream>>>(
      ao, 512, 0, OutW, 512, 0, opb, 0, 1,
      nullptr, out + 256, 768, 0, kSQ, 512, 512);
}
